// MultiHeadAttention_9045201125376
// MI455X (gfx1250) — compile-verified, matrix operand order checked by reading, not run on hardware
//
#include <hip/hip_runtime.h>

// ---------------------------------------------------------------------------
// MultiHeadAttention for MI455X (gfx1250): flash-attention with bf16 WMMA
// (v_wmma_f32_16x16x32_bf16), double-buffered GLOBAL_LOAD_ASYNC_TO_LDS
// prefetch of K/V chunks, packed bf16 conversion, + bf16 WMMA projection.
// ---------------------------------------------------------------------------

#define SEQ   2048
#define EMB   2048
#define NHEAD 16
#define HD    128
#define NB    2
#define BK    32            // keys per streamed chunk
#define NCHUNK (SEQ / BK)   // 64

typedef __bf16          v16bf __attribute__((ext_vector_type(16)));
typedef __bf16          v2bf  __attribute__((ext_vector_type(2)));
typedef float           v8f   __attribute__((ext_vector_type(8)));
typedef int             v4i   __attribute__((ext_vector_type(4)));
typedef unsigned short  u16;
typedef unsigned short  u16x8  __attribute__((ext_vector_type(8)));
typedef unsigned short  u16x16 __attribute__((ext_vector_type(16)));

// ---------------- scalar / packed f32 -> bf16 conversion -------------------
__device__ __forceinline__ u16 to_bf16(float f) {
  unsigned u = __builtin_bit_cast(unsigned, f);
  u += 0x7fffu + ((u >> 16) & 1u);           // round-to-nearest-even
  return (u16)(u >> 16);
}

__device__ __forceinline__ unsigned pack2_bf16(float a, float b) {
#if __has_builtin(__builtin_amdgcn_cvt_pk_bf16_f32)
  v2bf v = __builtin_amdgcn_cvt_pk_bf16_f32(a, b);
  return __builtin_bit_cast(unsigned, v);
#else
  return (unsigned)to_bf16(a) | ((unsigned)to_bf16(b) << 16);
#endif
}

// ---------------- async global -> LDS copy (16B per lane) ------------------
#if __has_builtin(__builtin_amdgcn_global_load_async_to_lds_b128)
#define HAVE_ASYNC 1
#else
#define HAVE_ASYNC 0
#endif

__device__ __forceinline__ void async_cp16(const float* g, float* l) {
#if HAVE_ASYNC
  typedef __attribute__((address_space(1))) v4i gv4i;
  typedef __attribute__((address_space(3))) v4i lv4i;
  __builtin_amdgcn_global_load_async_to_lds_b128((gv4i*)g, (lv4i*)l, 0, 0);
#else
  *(float4*)l = *(const float4*)g;     // synchronous fallback
#endif
}

__device__ __forceinline__ void wait_async_zero() {
#if HAVE_ASYNC
#if __has_builtin(__builtin_amdgcn_s_wait_asynccnt)
  __builtin_amdgcn_s_wait_asynccnt(0);
#else
  asm volatile("s_wait_asynccnt 0x0" ::: "memory");
#endif
#endif
}

__device__ __forceinline__ void wait_async_prev() {   // <= 16 outstanding
#if HAVE_ASYNC
#if __has_builtin(__builtin_amdgcn_s_wait_asynccnt)
  __builtin_amdgcn_s_wait_asynccnt(16);
#else
  asm volatile("s_wait_asynccnt 0x10" ::: "memory");
#endif
#endif
}

// ---------------- fragment helpers -----------------------------------------
__device__ __forceinline__ u16x16 pack_lds16(const u16* p) {
  u16x8 lo = *(const u16x8*)(p);
  u16x8 hi = *(const u16x8*)(p + 16);
  return __builtin_shufflevector(lo, hi, 0,1,2,3,4,5,6,7,8,9,10,11,12,13,14,15);
}

__device__ __forceinline__ u16x16 pack_f32x16(const float* p0, const float* p1) {
  const float4 a0 = *(const float4*)(p0);
  const float4 a1 = *(const float4*)(p0 + 4);
  const float4 b0 = *(const float4*)(p1);
  const float4 b1 = *(const float4*)(p1 + 4);
  u16x16 r;
  r[0]=to_bf16(a0.x);  r[1]=to_bf16(a0.y);  r[2]=to_bf16(a0.z);  r[3]=to_bf16(a0.w);
  r[4]=to_bf16(a1.x);  r[5]=to_bf16(a1.y);  r[6]=to_bf16(a1.z);  r[7]=to_bf16(a1.w);
  r[8]=to_bf16(b0.x);  r[9]=to_bf16(b0.y);  r[10]=to_bf16(b0.z); r[11]=to_bf16(b0.w);
  r[12]=to_bf16(b1.x); r[13]=to_bf16(b1.y); r[14]=to_bf16(b1.z); r[15]=to_bf16(b1.w);
  return r;
}

__device__ __forceinline__ v8f wmma_bf16(u16x16 a, u16x16 b, v8f c) {
  return __builtin_amdgcn_wmma_f32_16x16x32_bf16(
      false, __builtin_bit_cast(v16bf, a),
      false, __builtin_bit_cast(v16bf, b),
      (short)0, c, false, false);
}

// ---------------------------------------------------------------------------
// Kernel 1: flash attention. Block = 128 threads (4 waves); each wave owns a
// 16-row Q tile. K/V chunks async-prefetched one chunk ahead into double-
// buffered f32 LDS staging, converted to padded bf16 tiles, consumed via
// ds_load_b128 fragment gathers feeding v_wmma_f32_16x16x32_bf16.
// ---------------------------------------------------------------------------
__global__ __launch_bounds__(128) void attn_fwd_kernel(
    const float* __restrict__ values, const float* __restrict__ keys,
    const float* __restrict__ query,  const int* __restrict__ mask,
    u16* __restrict__ Obuf)
{
  __shared__ float Kst[2][BK][HD];     // async staging, double buffered
  __shared__ float Vst[2][BK][HD];     // async staging, double buffered
  __shared__ u16 Klds[BK][HD + 8];     // key-major bf16, padded rows
  __shared__ u16 Vlds[HD][BK + 8];     // d-major (transposed) bf16, padded
  __shared__ u16 Plds[4][16][BK];      // per-wave P staging (C -> A layout)

  const int tid  = threadIdx.x;
  const int wave = tid >> 5;
  const int lane = tid & 31;
  const int ln   = lane & 15;
  const int kh   = lane >> 4;
  const int h    = blockIdx.y;
  const int b    = blockIdx.z;
  const int qbase = blockIdx.x * 64 + wave * 16;

  const size_t bh_off = (size_t)b * SEQ * EMB + (size_t)h * HD;
  const float* Qp = query  + bh_off + (size_t)qbase * EMB;
  const float* Kp = keys   + bh_off;
  const float* Vp = values + bh_off;
  const int*   Mp = mask + (size_t)b * SEQ * SEQ;

  u16x16 qa[4];
#pragma unroll
  for (int dk = 0; dk < 4; ++dk) {
    const float* r = Qp + (size_t)ln * EMB + dk * 32 + kh * 8;
    qa[dk] = pack_f32x16(r, r + 16);
  }

  v8f o[8];
  float m[8], l[8];
#pragma unroll
  for (int i = 0; i < 8; ++i) { o[i] = (v8f)0.0f; m[i] = -3.0e38f; l[i] = 0.0f; }

  const float scale = 0.022097086912079608f;   // 1/sqrt(EMB)

  auto issue_chunk = [&](int kc, int buf) {
#pragma unroll
    for (int i = 0; i < 8; ++i) {
      const int u   = tid + i * 128;
      const int key = u >> 5;
      const int dq  = (u & 31) * 4;
      async_cp16(Kp + (size_t)(kc + key) * EMB + dq, &Kst[buf][key][dq]);
    }
#pragma unroll
    for (int i = 0; i < 8; ++i) {
      const int u   = tid + i * 128;
      const int key = u >> 5;
      const int dq  = (u & 31) * 4;
      async_cp16(Vp + (size_t)(kc + key) * EMB + dq, &Vst[buf][key][dq]);
    }
  };

  issue_chunk(0, 0);                       // prologue prefetch

  for (int ic = 0; ic < NCHUNK; ++ic) {
    const int kc  = ic * BK;
    const int buf = ic & 1;

    if (ic + 1 < NCHUNK) { issue_chunk(kc + BK, buf ^ 1); wait_async_prev(); }
    else                 { wait_async_zero(); }
    __syncthreads();

    // ---- convert pass: f32 staging -> padded bf16 tiles ----
#pragma unroll
    for (int i = 0; i < 8; ++i) {
      const int u   = tid + i * 128;
      const int key = u >> 5;
      const int dq  = (u & 31) * 4;
      const float4 v = *(const float4*)&Kst[buf][key][dq];
      *(unsigned*)&Klds[key][dq]     = pack2_bf16(v.x, v.y);
      *(unsigned*)&Klds[key][dq + 2] = pack2_bf16(v.z, v.w);
    }
    {
      const int d = tid;
#pragma unroll
      for (int i = 0; i < 16; ++i) {
        const float a0 = Vst[buf][2 * i][d];
        const float a1 = Vst[buf][2 * i + 1][d];
        *(unsigned*)&Vlds[d][2 * i] = pack2_bf16(a0, a1);
      }
    }
    __syncthreads();

    // ---- S = Q K^T ----
    v8f s[2];
#pragma unroll
    for (int j = 0; j < 2; ++j) {
      v8f acc = (v8f)0.0f;
      const u16* kr = &Klds[j * 16 + ln][kh * 8];
#pragma unroll
      for (int dk = 0; dk < 4; ++dk)
        acc = wmma_bf16(qa[dk], pack_lds16(kr + dk * 32), acc);
      s[j] = acc;
    }

    // ---- online softmax + P store ----
#pragma unroll
    for (int r = 0; r < 8; ++r) {
      const int qrow = qbase + r + 8 * kh;
      float e0 = s[0][r] * scale;
      float e1 = s[1][r] * scale;
      if (Mp[(size_t)qrow * SEQ + kc + ln] == 0)      e0 = -1.0e30f;
      if (Mp[(size_t)qrow * SEQ + kc + 16 + ln] == 0) e1 = -1.0e30f;
      float t = fmaxf(e0, e1);
#pragma unroll
      for (int x = 8; x >= 1; x >>= 1) t = fmaxf(t, __shfl_xor(t, x, 32));
      const float mn = fmaxf(m[r], t);
      const float c  = __expf(m[r] - mn);
      const float p0 = __expf(e0 - mn);
      const float p1 = __expf(e1 - mn);
      float rs = p0 + p1;
#pragma unroll
      for (int x = 8; x >= 1; x >>= 1) rs += __shfl_xor(rs, x, 32);
      l[r] = l[r] * c + rs;
      m[r] = mn;
#pragma unroll
      for (int dt = 0; dt < 8; ++dt) o[dt][r] *= c;
      Plds[wave][r + 8 * kh][ln]      = to_bf16(p0);
      Plds[wave][r + 8 * kh][16 + ln] = to_bf16(p1);
    }
    __syncthreads();

    // ---- O += P V ----
    const u16x16 pa = pack_lds16(&Plds[wave][ln][kh * 8]);
#pragma unroll
    for (int dt = 0; dt < 8; ++dt)
      o[dt] = wmma_bf16(pa, pack_lds16(&Vlds[dt * 16 + ln][kh * 8]), o[dt]);
  }

#pragma unroll
  for (int r = 0; r < 8; ++r) {
    const int qrow = qbase + r + 8 * kh;
    const float inv = 1.0f / l[r];
    const size_t base = ((size_t)b * SEQ + qrow) * EMB + (size_t)h * HD + ln;
#pragma unroll
    for (int dt = 0; dt < 8; ++dt)
      Obuf[base + dt * 16] = to_bf16(o[dt][r] * inv);
  }
}

// ---------------------------------------------------------------------------
// Kernel 2: out = O @ W^T + b, bf16 WMMA, W staged in LDS per 64x64 tile.
// ---------------------------------------------------------------------------
__global__ __launch_bounds__(128) void proj_kernel(
    const u16* __restrict__ Obuf, const float* __restrict__ W,
    const float* __restrict__ bias, float* __restrict__ out)
{
  __shared__ u16 Wlds[64][40];

  const int tid  = threadIdx.x;
  const int wave = tid >> 5;
  const int lane = tid & 31;
  const int ln   = lane & 15;
  const int kh   = lane >> 4;
  const int eo0  = blockIdx.x * 64;
  const int row0 = blockIdx.y * 64;
  const int myrow = row0 + wave * 16 + ln;

  v8f acc[4];
#pragma unroll
  for (int i = 0; i < 4; ++i) acc[i] = (v8f)0.0f;

  for (int ei = 0; ei < EMB; ei += 32) {
    __syncthreads();
    {
      const int eol = tid >> 1;
      const int seg = (tid & 1) * 16;
      const float* src = W + (size_t)(eo0 + eol) * EMB + ei + seg;
#pragma unroll
      for (int i = 0; i < 4; ++i) {
        const float4 v = *(const float4*)(src + 4 * i);
        *(unsigned*)&Wlds[eol][seg + 4 * i]     = pack2_bf16(v.x, v.y);
        *(unsigned*)&Wlds[eol][seg + 4 * i + 2] = pack2_bf16(v.z, v.w);
      }
    }
    __syncthreads();

    const u16x16 aO = pack_lds16(Obuf + (size_t)myrow * EMB + ei + kh * 8);
#pragma unroll
    for (int et = 0; et < 4; ++et)
      acc[et] = wmma_bf16(aO, pack_lds16(&Wlds[et * 16 + ln][kh * 8]), acc[et]);
  }

#pragma unroll
  for (int et = 0; et < 4; ++et) {
    const int eo = eo0 + et * 16 + ln;
    const float bv = bias[eo];
#pragma unroll
    for (int r = 0; r < 8; ++r) {
      const int row = row0 + wave * 16 + r + 8 * kh;
      out[(size_t)row * EMB + eo] = acc[et][r] + bv;
    }
  }
}

// ---------------------------------------------------------------------------
extern "C" void kernel_launch(void* const* d_in, const int* in_sizes, int n_in,
                              void* d_out, int out_size, void* d_ws, size_t ws_size,
                              hipStream_t stream) {
  const float* values = (const float*)d_in[0];
  const float* keys   = (const float*)d_in[1];
  const float* query  = (const float*)d_in[2];
  const int*   mask   = (const int*)  d_in[3];
  const float* W      = (const float*)d_in[4];
  const float* bias   = (const float*)d_in[5];
  float* out = (float*)d_out;
  u16*   Obuf = (u16*)d_ws;   // N*SEQ x EMB bf16 intermediate (16.8 MB)

  dim3 g1(SEQ / 64, NHEAD, NB);
  attn_fwd_kernel<<<g1, 128, 0, stream>>>(values, keys, query, mask, Obuf);

  dim3 g2(EMB / 64, (NB * SEQ) / 64, 1);
  proj_kernel<<<g2, 128, 0, stream>>>(Obuf, W, bias, out);
}
